// Transformer_57114475102271
// MI455X (gfx1250) — matrix operand order repaired and checked by reading, not compiled, not run on hardware
//
#include <hip/hip_runtime.h>
#include <hip/hip_bf16.h>

// ---------------- problem constants ----------------
constexpr int kL  = 6;
constexpr int kH  = 512;
constexpr int kNH = 8;
constexpr int kDK = 64;      // kH / kNH
constexpr int kFF = 2048;
constexpr int kV  = 10000;
constexpr int kVp = 10048;   // kV padded to multiple of 64 (k_wcvt tile)
constexpr int kB  = 4;
constexpr int kS  = 2048;
constexpr int kM  = kB * kS; // 8192 rows

// ---------------- WMMA types ----------------
typedef __attribute__((ext_vector_type(16))) __bf16 v16bf;
typedef __attribute__((ext_vector_type(8)))  float  v8f;

union FragU  { uint4 q[2]; v16bf v; };
union Chunk8 { uint4 u; __bf16 e[8]; };

static __device__ __forceinline__ __bf16 f2bf(float f) {
  // round-to-nearest-even fp32 -> bf16
  unsigned u = __float_as_uint(f);
  unsigned r = (u + 0x7FFFu + ((u >> 16) & 1u)) >> 16;
  unsigned short s = (unsigned short)r;
  return __builtin_bit_cast(__bf16, s);
}

static __device__ __forceinline__ v8f wmma_bf16(v16bf a, v16bf b, v8f c) {
  return __builtin_amdgcn_wmma_f32_16x16x32_bf16(false, a, false, b, (short)0, c,
                                                 false, false);
}

// CDNA5 async global->LDS copy (16B per lane), tracked with ASYNCcnt.
static __device__ __forceinline__ void async_cp16(void* lds, const void* gptr) {
  unsigned loff = (unsigned)(size_t)lds;   // generic addr low 32 bits = LDS offset
  asm volatile("global_load_async_to_lds_b128 %0, %1, off"
               :: "v"(loff), "v"(gptr) : "memory");
}
static __device__ __forceinline__ void async_wait0() {
  asm volatile("s_wait_asynccnt 0x0" ::: "memory");
}

// ---------------- embedding + positional encoding (fp32 residual x) --------
__global__ __launch_bounds__(256) void k_embed(const int* __restrict__ src,
                                               const float* __restrict__ emb,
                                               float* __restrict__ x) {
  int row = blockIdx.x;            // 0..kM-1
  int s   = row % kS;
  int tok = src[row];
  const float c0 = -9.210340371976184f / (float)kH;  // -ln(10000)/H
  for (int c = threadIdx.x; c < kH; c += 256) {
    int p2 = c & ~1;
    float ang = (float)s * __expf((float)p2 * c0);
    float pe  = (c & 1) ? __cosf(ang) : __sinf(ang);
    x[(size_t)row * kH + c] = emb[(size_t)tok * kH + c] + pe;
  }
}

// ---- weight convert + transpose: W[K][N] f32 -> Wt[Npad][K] bf16 (0-padded)
__global__ __launch_bounds__(256) void k_wcvt(const float* __restrict__ W,
                                              __bf16* __restrict__ Wt,
                                              int K, int N) {
  __shared__ float T[64][65];
  const int n0 = blockIdx.x * 64;           // grid.x covers Npad/64
  const int k0 = blockIdx.y * 64;
  for (int i = 0; i < 16; ++i) {
    int idx = threadIdx.x + i * 256;        // 0..4095
    int r = idx >> 6, c = idx & 63;         // r = k-local, c = n-local
    int gn = n0 + c;
    T[r][c] = (gn < N) ? W[(size_t)(k0 + r) * N + gn] : 0.f;
  }
  __syncthreads();
  for (int i = 0; i < 16; ++i) {
    int idx = threadIdx.x + i * 256;
    int r = idx >> 6, c = idx & 63;         // r = n-local, c = k-local
    Wt[(size_t)(n0 + r) * K + k0 + c] = f2bf(T[c][r]);  // pad rows get zeros
  }
}

// ---------------- layernorm: fp32 in -> bf16 out ----------------
__global__ __launch_bounds__(256) void k_layernorm(const float* __restrict__ x,
                                                   const float* __restrict__ g,
                                                   const float* __restrict__ b,
                                                   __bf16* __restrict__ y) {
  __shared__ float r1[256], r2[256];
  int row = blockIdx.x;
  const float* xr = x + (size_t)row * kH;
  float s = 0.f, s2 = 0.f;
  for (int c = threadIdx.x; c < kH; c += 256) {
    float v = xr[c]; s += v; s2 += v * v;
  }
  r1[threadIdx.x] = s; r2[threadIdx.x] = s2;
  __syncthreads();
  for (int st = 128; st > 0; st >>= 1) {
    if (threadIdx.x < st) { r1[threadIdx.x] += r1[threadIdx.x + st];
                            r2[threadIdx.x] += r2[threadIdx.x + st]; }
    __syncthreads();
  }
  float m   = r1[0] / (float)kH;
  float var = r2[0] / (float)kH - m * m;
  float inv = rsqrtf(var + 1e-5f);
  for (int c = threadIdx.x; c < kH; c += 256)
    y[(size_t)row * kH + c] = f2bf((xr[c] - m) * inv * g[c] + b[c]);
}

// ---------------- bf16-WMMA GEMM, 128x128x32 tile, double-buffered ----------
// C[M,N] = act(A[M,K]bf16 @ Wt[Npad,K]^T bf16 + bias) (+ residual fp32)
// grid = (Npad/128, M/128), block = 256 (8 waves: 4(M) x 2(N))
__global__ __launch_bounds__(256) void k_gemm(const __bf16* __restrict__ A,
                                              const __bf16* __restrict__ Wt,
                                              const float* __restrict__ bias,
                                              const float* __restrict__ res,
                                              void* __restrict__ Cout,
                                              int N, int K, int relu, int out_bf16) {
  __shared__ __bf16 As[2][128][32];   // [buf][m][k]
  __shared__ __bf16 Bs[2][128][32];   // [buf][n][k]
  const int tid  = threadIdx.x;
  const int lane = tid & 31;
  const int wid  = tid >> 5;
  const int l    = lane & 15;
  const int hi   = lane >> 4;
  const int wm   = wid >> 1;       // 0..3 -> 32-row slab
  const int wn   = wid & 1;        // 0..1 -> 64-col slab
  const int m0   = blockIdx.y * 128;
  const int n0   = blockIdx.x * 128;

  v8f acc[2][4];
  for (int mi = 0; mi < 2; ++mi)
    for (int ni = 0; ni < 4; ++ni) acc[mi][ni] = {};

  const int ar = tid >> 2, ac8 = (tid & 3) << 3;   // chunk coords (row, col8)

#define GSTAGE(buf, kt) do {                                                   \
    const size_t ko = (size_t)(kt) * 32 + ac8;                                 \
    async_cp16(&As[buf][ar][ac8],      A  + (size_t)(m0 + ar) * K + ko);       \
    async_cp16(&As[buf][ar + 64][ac8], A  + (size_t)(m0 + ar + 64) * K + ko);  \
    async_cp16(&Bs[buf][ar][ac8],      Wt + (size_t)(n0 + ar) * K + ko);       \
    async_cp16(&Bs[buf][ar + 64][ac8], Wt + (size_t)(n0 + ar + 64) * K + ko);  \
  } while (0)

#define GCOMP(buf) do {                                                        \
    FragU af[2], bfg[4];                                                       \
    for (int mi = 0; mi < 2; ++mi) {                                           \
      const __bf16* pA = &As[buf][wm * 32 + mi * 16 + l][0];                   \
      af[mi].q[0] = *(const uint4*)(pA + hi * 8);                              \
      af[mi].q[1] = *(const uint4*)(pA + 16 + hi * 8);                         \
    }                                                                          \
    for (int ni = 0; ni < 4; ++ni) {                                           \
      const __bf16* pB = &Bs[buf][wn * 64 + ni * 16 + l][hi * 8];              \
      bfg[ni].q[0] = *(const uint4*)(pB);                                      \
      bfg[ni].q[1] = *(const uint4*)(pB + 16);                                 \
    }                                                                          \
    for (int mi = 0; mi < 2; ++mi)                                             \
      for (int ni = 0; ni < 4; ++ni)                                           \
        acc[mi][ni] = wmma_bf16(af[mi].v, bfg[ni].v, acc[mi][ni]);             \
  } while (0)

  const int nkt = K >> 5;          // always even (16 or 64)
  GSTAGE(0, 0);
  async_wait0();
  __syncthreads();
  for (int kt = 0; kt < nkt; kt += 2) {
    GSTAGE(1, kt + 1);             // DMA next tile while computing current
    GCOMP(0);
    async_wait0();
    __syncthreads();
    if (kt + 2 < nkt) GSTAGE(0, kt + 2);
    GCOMP(1);
    async_wait0();
    __syncthreads();
  }
#undef GSTAGE
#undef GCOMP

  // epilogue: bias (+residual) (+relu), N-guarded coalesced stores
  for (int mi = 0; mi < 2; ++mi)
    for (int ni = 0; ni < 4; ++ni)
      for (int r = 0; r < 8; ++r) {
        int rl = wm * 32 + mi * 16 + hi * 8 + r;
        int cl = wn * 64 + ni * 16 + l;
        int gr = m0 + rl, gc = n0 + cl;
        if (gc < N) {
          float v = acc[mi][ni][r] + bias[gc];
          if (res)  v += res[(size_t)gr * N + gc];
          if (relu) v = fmaxf(v, 0.f);
          if (out_bf16) ((__bf16*)Cout)[(size_t)gr * N + gc] = f2bf(v);
          else          ((float*)Cout)[(size_t)gr * N + gc] = v;
        }
      }
}

// ---------------- fused flash attention (bf16 in / bf16 out) ----------------
// grid = (S/64 query blocks, B*NH), block = 256 (8 waves, 4x2)
__global__ __launch_bounds__(256) void k_attn(const __bf16* __restrict__ q,
                                              const __bf16* __restrict__ k,
                                              const __bf16* __restrict__ v,
                                              const unsigned char* __restrict__ mask,
                                              __bf16* __restrict__ ctx) {
  __shared__ __bf16 Qs[64][64];          // [query][dk]   (A layout)
  __shared__ __bf16 Ks[64][64];          // [key][dk]     (B layout for Q@K^T)
  __shared__ __bf16 Vs[64][64];          // [dk][key]     (B layout for P@V)
  __shared__ __bf16 Ps[64][64];          // [query][key]  (A layout)
  __shared__ float  m_s[64], l_s[64], alpha_s[64], moff_s[64];
  __shared__ float  red_s[2][64];

  const int tid  = threadIdx.x;
  const int lane = tid & 31;
  const int wid  = tid >> 5;
  const int l    = lane & 15;
  const int hi   = lane >> 4;
  const int wm   = wid >> 1;             // 16-row slab of queries
  const int wn   = wid & 1;              // 32-col slab (keys / dk)
  const int qb   = blockIdx.x;
  const int bh   = blockIdx.y;
  const int b    = bh / kNH;
  const int head = bh % kNH;
  const size_t base = (size_t)b * kS * kH + (size_t)head * kDK;

  const int cr = tid >> 2, cc8 = (tid & 3) << 3;   // 64x64: 512 chunks, 2/thread

  // stage Q (pure async copy; scale folded into score epilogue)
  async_cp16(&Qs[cr][cc8],      q + base + (size_t)(qb * 64 + cr) * kH + cc8);
  async_cp16(&Qs[cr + 16][cc8], q + base + (size_t)(qb * 64 + cr + 16) * kH + cc8);
  if (tid < 64) { m_s[tid] = -1e30f; l_s[tid] = 0.f; }
  async_wait0();
  __syncthreads();

  FragU qf[2];
  for (int ks = 0; ks < 2; ++ks) {
    const __bf16* p = &Qs[wm * 16 + l][ks * 32];
    qf[ks].q[0] = *(const uint4*)(p + hi * 8);
    qf[ks].q[1] = *(const uint4*)(p + 16 + hi * 8);
  }
  v8f o0 = {}, o1 = {};

  for (int kb = 0; kb < kS / 64; ++kb) {
    // stage K (async pure copy) + V (batched sync loads, LDS transpose)
    async_cp16(&Ks[cr][cc8],      k + base + (size_t)(kb * 64 + cr) * kH + cc8);
    async_cp16(&Ks[cr + 16][cc8], k + base + (size_t)(kb * 64 + cr + 16) * kH + cc8);
    Chunk8 v0c, v1c;
    v0c.u = *(const uint4*)(v + base + (size_t)(kb * 64 + cr) * kH + cc8);
    v1c.u = *(const uint4*)(v + base + (size_t)(kb * 64 + cr + 16) * kH + cc8);
    for (int j = 0; j < 8; ++j) {
      Vs[cc8 + j][cr]      = v0c.e[j];
      Vs[cc8 + j][cr + 16] = v1c.e[j];
    }
    if (tid < 64)
      moff_s[tid] = mask[(size_t)b * kS + kb * 64 + tid] ? -1e30f : 0.f;
    async_wait0();
    __syncthreads();

    // scores: S = Q @ K^T over dk (2 k-steps of 32)
    v8f s0 = {}, s1 = {};
    for (int ks = 0; ks < 2; ++ks) {
      FragU k0, k1;
      const __bf16* p0 = &Ks[wn * 32 + l][ks * 32 + hi * 8];
      k0.q[0] = *(const uint4*)(p0); k0.q[1] = *(const uint4*)(p0 + 16);
      const __bf16* p1 = &Ks[wn * 32 + 16 + l][ks * 32 + hi * 8];
      k1.q[0] = *(const uint4*)(p1); k1.q[1] = *(const uint4*)(p1 + 16);
      s0 = wmma_bf16(qf[ks].v, k0.v, s0);
      s1 = wmma_bf16(qf[ks].v, k1.v, s1);
    }
    // scale 1/sqrt(DK) + mask
    float mo0 = moff_s[wn * 32 + l], mo1 = moff_s[wn * 32 + 16 + l];
    for (int r = 0; r < 8; ++r) {
      s0[r] = s0[r] * 0.125f + mo0;
      s1[r] = s1[r] * 0.125f + mo1;
    }

    // per-row block max (16-lane butterfly + cross-wave via LDS)
    for (int r = 0; r < 8; ++r) {
      float rv = fmaxf(s0[r], s1[r]);
      for (int off = 1; off < 16; off <<= 1)
        rv = fmaxf(rv, __shfl_xor(rv, off, 32));
      if (l == 0) red_s[wn][wm * 16 + hi * 8 + r] = rv;
    }
    __syncthreads();
    if (tid < 64) {
      float mb = fmaxf(red_s[0][tid], red_s[1][tid]);
      float mo = m_s[tid];
      float mn = fmaxf(mo, mb);
      alpha_s[tid] = __expf(mo - mn);
      m_s[tid] = mn;
    }
    __syncthreads();

    // P = exp(S - m), rowsum, write P to LDS (bf16)
    for (int r = 0; r < 8; ++r) {
      int row = wm * 16 + hi * 8 + r;
      float mn = m_s[row];
      float p0 = __expf(s0[r] - mn);
      float p1 = __expf(s1[r] - mn);
      Ps[row][wn * 32 + l]      = f2bf(p0);
      Ps[row][wn * 32 + 16 + l] = f2bf(p1);
      float rs = p0 + p1;
      for (int off = 1; off < 16; off <<= 1)
        rs += __shfl_xor(rs, off, 32);
      if (l == 0) red_s[wn][row] = rs;
    }
    __syncthreads();
    if (tid < 64)
      l_s[tid] = l_s[tid] * alpha_s[tid] + red_s[0][tid] + red_s[1][tid];

    // O = O*alpha + P @ V
    for (int r = 0; r < 8; ++r) {
      float a = alpha_s[wm * 16 + hi * 8 + r];
      o0[r] *= a; o1[r] *= a;
    }
    for (int ks = 0; ks < 2; ++ks) {
      FragU pf, v0f, v1f;
      const __bf16* pp = &Ps[wm * 16 + l][ks * 32];
      pf.q[0] = *(const uint4*)(pp + hi * 8);
      pf.q[1] = *(const uint4*)(pp + 16 + hi * 8);
      const __bf16* pv0 = &Vs[wn * 32 + l][ks * 32 + hi * 8];
      v0f.q[0] = *(const uint4*)(pv0); v0f.q[1] = *(const uint4*)(pv0 + 16);
      const __bf16* pv1 = &Vs[wn * 32 + 16 + l][ks * 32 + hi * 8];
      v1f.q[0] = *(const uint4*)(pv1); v1f.q[1] = *(const uint4*)(pv1 + 16);
      o0 = wmma_bf16(pf.v, v0f.v, o0);
      o1 = wmma_bf16(pf.v, v1f.v, o1);
    }
    __syncthreads();
  }

  // finalize: O / l, scatter to ctx in [b, s, head*DK + dk] layout (bf16)
  for (int r = 0; r < 8; ++r) {
    int row = wm * 16 + hi * 8 + r;
    float inv = 1.f / l_s[row];
    size_t ob = base + (size_t)(qb * 64 + row) * kH;
    ctx[ob + wn * 32 + l]      = f2bf(o0[r] * inv);
    ctx[ob + wn * 32 + 16 + l] = f2bf(o1[r] * inv);
  }
}

// ---------------- host orchestration ----------------
extern "C" void kernel_launch(void* const* d_in, const int* in_sizes, int n_in,
                              void* d_out, int out_size, void* d_ws, size_t ws_size,
                              hipStream_t stream) {
  const int*           src  = (const int*)d_in[0];
  const unsigned char* mask = (const unsigned char*)d_in[1];
  const float* emb  = (const float*)d_in[2];
  const float* Wq   = (const float*)d_in[3];
  const float* bq   = (const float*)d_in[4];
  const float* Wk   = (const float*)d_in[5];
  const float* bk   = (const float*)d_in[6];
  const float* Wv   = (const float*)d_in[7];
  const float* bv   = (const float*)d_in[8];
  const float* Wo   = (const float*)d_in[9];
  const float* bo   = (const float*)d_in[10];
  const float* ln1g = (const float*)d_in[11];
  const float* ln1b = (const float*)d_in[12];
  const float* W1   = (const float*)d_in[13];
  const float* b1   = (const float*)d_in[14];
  const float* W2   = (const float*)d_in[15];
  const float* b2   = (const float*)d_in[16];
  const float* ln2g = (const float*)d_in[17];
  const float* ln2b = (const float*)d_in[18];
  const float* lnfg = (const float*)d_in[19];
  const float* lnfb = (const float*)d_in[20];
  const float* Wg   = (const float*)d_in[21];
  const float* bg   = (const float*)d_in[22];
  float* out = (float*)d_out;

  // ---- workspace carve-up ----
  char* p = (char*)d_ws;
  float* x = (float*)p;            p += (size_t)kM * kH * 4;   // fp32 residual
  __bf16* h   = (__bf16*)p;        p += (size_t)kM * kH * 2;
  __bf16* qb_ = (__bf16*)p;        p += (size_t)kM * kH * 2;
  __bf16* kb_ = (__bf16*)p;        p += (size_t)kM * kH * 2;
  __bf16* vb_ = (__bf16*)p;        p += (size_t)kM * kH * 2;
  __bf16* cb_ = (__bf16*)p;        p += (size_t)kM * kH * 2;
  __bf16* ffn = (__bf16*)p;        p += (size_t)kM * kFF * 2;
  __bf16* wqT = (__bf16*)p;        p += (size_t)kL * kH * kH * 2;
  __bf16* wkT = (__bf16*)p;        p += (size_t)kL * kH * kH * 2;
  __bf16* wvT = (__bf16*)p;        p += (size_t)kL * kH * kH * 2;
  __bf16* woT = (__bf16*)p;        p += (size_t)kL * kH * kH * 2;
  __bf16* w1T = (__bf16*)p;        p += (size_t)kL * kH * kFF * 2;
  __bf16* w2T = (__bf16*)p;        p += (size_t)kL * kFF * kH * 2;
  __bf16* wgT = (__bf16*)p;        p += (size_t)kVp * kH * 2;   // padded rows

  const dim3 blk(256);
  const dim3 tHH(kH / 64, kH / 64);            // transpose grids
  const dim3 tHF(kFF / 64, kH / 64);
  const dim3 tFH(kH / 64, kFF / 64);
  const dim3 tHV(kVp / 64, kH / 64);           // covers padded vocab rows
  const dim3 gH(kH / 128, kM / 128);           // gemm grids
  const dim3 gF(kFF / 128, kM / 128);
  const dim3 gV(kVp / 128, kM / 128);

  // ---- one-time weight convert+transpose to bf16 ----
  for (int li = 0; li < kL; ++li) {
    size_t oHH = (size_t)li * kH * kH, oHF = (size_t)li * kH * kFF;
    k_wcvt<<<tHH, blk, 0, stream>>>(Wq + oHH, wqT + oHH, kH, kH);
    k_wcvt<<<tHH, blk, 0, stream>>>(Wk + oHH, wkT + oHH, kH, kH);
    k_wcvt<<<tHH, blk, 0, stream>>>(Wv + oHH, wvT + oHH, kH, kH);
    k_wcvt<<<tHH, blk, 0, stream>>>(Wo + oHH, woT + oHH, kH, kH);
    k_wcvt<<<tHF, blk, 0, stream>>>(W1 + oHF, w1T + oHF, kH, kFF);
    k_wcvt<<<tFH, blk, 0, stream>>>(W2 + oHF, w2T + oHF, kFF, kH);
  }
  k_wcvt<<<tHV, blk, 0, stream>>>(Wg, wgT, kH, kV);

  k_embed<<<dim3(kM), blk, 0, stream>>>(src, emb, x);

  for (int li = 0; li < kL; ++li) {
    size_t oHH = (size_t)li * kH * kH, oHF = (size_t)li * kH * kFF;
    k_layernorm<<<dim3(kM), blk, 0, stream>>>(x, ln1g + li * kH, ln1b + li * kH, h);
    k_gemm<<<gH, blk, 0, stream>>>(h, wqT + oHH, bq + li * kH, nullptr, qb_,
                                   kH, kH, 0, 1);
    k_gemm<<<gH, blk, 0, stream>>>(h, wkT + oHH, bk + li * kH, nullptr, kb_,
                                   kH, kH, 0, 1);
    k_gemm<<<gH, blk, 0, stream>>>(h, wvT + oHH, bv + li * kH, nullptr, vb_,
                                   kH, kH, 0, 1);
    k_attn<<<dim3(kS / 64, kB * kNH), blk, 0, stream>>>(qb_, kb_, vb_, mask, cb_);
    k_gemm<<<gH, blk, 0, stream>>>(cb_, woT + oHH, bo + li * kH, x, x,
                                   kH, kH, 0, 0);
    k_layernorm<<<dim3(kM), blk, 0, stream>>>(x, ln2g + li * kH, ln2b + li * kH, h);
    k_gemm<<<gF, blk, 0, stream>>>(h, w1T + oHF, b1 + li * kFF, nullptr, ffn,
                                   kFF, kH, 1, 1);
    k_gemm<<<gH, blk, 0, stream>>>(ffn, w2T + oHF, b2 + li * kH, x, x,
                                   kH, kFF, 0, 0);
  }

  k_layernorm<<<dim3(kM), blk, 0, stream>>>(x, lnfg, lnfb, h);
  k_gemm<<<gV, blk, 0, stream>>>(h, wgT, bg, nullptr, out, kV, kH, 0, 0);
}
